// MultiHeadAttention_61366492725743
// MI455X (gfx1250) — matrix operand order repaired and checked by reading, not compiled, not run on hardware
//
#include <hip/hip_runtime.h>
#include <hip/hip_bf16.h>

// Problem constants (match the reference).
constexpr int BB = 2;       // batch
constexpr int SS = 2048;    // sequence
constexpr int NH = 16;      // heads
constexpr int HD = 64;      // head dim
constexpr int DD = 1024;    // hidden
constexpr float LOG2E = 1.4426950408889634f;

typedef __attribute__((ext_vector_type(16))) __bf16 v16bf;
typedef __attribute__((ext_vector_type(8)))  float  v8f;

union BFrag {                 // one WMMA 16-bit operand fragment (16 halves / lane)
    v16bf v;
    uint4 q[2];
    __bf16 e[16];
};

__device__ inline void cvt8(const float4& x, const float4& y, __bf16* d) {
    d[0] = (__bf16)x.x; d[1] = (__bf16)x.y; d[2] = (__bf16)x.z; d[3] = (__bf16)x.w;
    d[4] = (__bf16)y.x; d[5] = (__bf16)y.y; d[6] = (__bf16)y.z; d[7] = (__bf16)y.w;
}

__device__ inline v8f wmma_bf16(const v16bf& a, const v16bf& b, const v8f& c) {
    return __builtin_amdgcn_wmma_f32_16x16x32_bf16(
        /*neg_a=*/false, a, /*neg_b=*/false, b,
        /*c_mod=*/(short)0, c, /*reuse_a=*/false, /*reuse_b=*/false);
}

// ---- DPP 16-lane reductions (pure VALU, no LDS crossbar) -------------------
// quad_perm xor1 = 0xB1, quad_perm xor2 = 0x4E,
// row_half_mirror = 0x141 (pairs quads within 8), row_mirror = 0x140 (pairs 8s within 16)
template <int CTRL>
__device__ inline float dpp_xchg(float x) {
    int r = __builtin_amdgcn_update_dpp(0, __builtin_bit_cast(int, x),
                                        CTRL, 0xf, 0xf, true);
    return __builtin_bit_cast(float, r);
}
__device__ inline float red_max16(float x) {
    x = fmaxf(x, dpp_xchg<0xB1>(x));
    x = fmaxf(x, dpp_xchg<0x4E>(x));
    x = fmaxf(x, dpp_xchg<0x141>(x));
    x = fmaxf(x, dpp_xchg<0x140>(x));
    return x;
}
__device__ inline float red_sum16(float x) {
    x += dpp_xchg<0xB1>(x);
    x += dpp_xchg<0x4E>(x);
    x += dpp_xchg<0x141>(x);
    x += dpp_xchg<0x140>(x);
    return x;
}

// ---------------------------------------------------------------------------
// Stage 1: projection GEMM  C[4096,1024] = A @ W^T + bias, output as bf16.
// MODE 0: Q  -> Qh[b][h][s][d], scaled by 1/sqrt(HD)=0.125 (exact in bf16)
// MODE 1: K  -> Kh[b][h][s][d]
// MODE 2: V  -> Vt[b][h][dv][s]   (transposed so P@V B-frags load contiguously)
// One wave = 16x64 output tile (4 N-subtiles of 16), k-loop step 32.
// ---------------------------------------------------------------------------
template <int MODE>
__launch_bounds__(256)
__global__ void proj_kernel(const float* __restrict__ A,   // [4096][1024]
                            const float* __restrict__ W,   // [1024][1024]
                            const float* __restrict__ bias,// [1024]
                            __bf16* __restrict__ O) {
    const int lane = threadIdx.x & 31;
    const int wv   = threadIdx.x >> 5;
    const int w    = blockIdx.x * 8 + wv;     // 4096 waves
    const int mt   = w >> 4;                  // 0..255 : 16-row tile of A
    const int ng   = w & 15;                  // 0..15  : 64-col group == head
    const int ln   = lane & 15;
    const int hi   = lane >> 4;

    const float* arow = A + (size_t)(mt * 16 + ln) * DD;

    v8f acc[4];
#pragma unroll
    for (int t = 0; t < 4; ++t) acc[t] = (v8f){0.f,0.f,0.f,0.f,0.f,0.f,0.f,0.f};

    for (int kb = 0; kb < DD; kb += 32) {
        // A fragment: lane holds K = {kb+hi*16 .. +15}
        BFrag af;
        {
            const float* p0 = arow + kb + hi * 16;
            const float* p1 = arow + kb + hi * 16 + 8;
            float4 f0 = *(const float4*)p0, f1 = *(const float4*)(p0 + 4);
            float4 f2 = *(const float4*)p1, f3 = *(const float4*)(p1 + 4);
            cvt8(f0, f1, af.e);
            cvt8(f2, f3, af.e + 8);
        }
#pragma unroll
        for (int nt = 0; nt < 4; ++nt) {
            // B fragment: lane = out column n, holds W[n][kb+hi*16 .. +15]
            const float* wp = W + (size_t)(ng * 64 + nt * 16 + ln) * DD + kb + hi * 16;
            float4 g0 = *(const float4*)wp,        g1 = *(const float4*)(wp + 4);
            float4 g2 = *(const float4*)(wp + 8),  g3 = *(const float4*)(wp + 12);
            BFrag bf;
            cvt8(g0, g1, bf.e);
            cvt8(g2, g3, bf.e + 8);
            acc[nt] = wmma_bf16(af.v, bf.v, acc[nt]);
        }
    }

    // bias + store (bf16) in the stage-2 friendly layout
    const int gm0 = mt * 16;
#pragma unroll
    for (int nt = 0; nt < 4; ++nt) {
        const int n  = ng * 64 + nt * 16 + ln;   // global feature
        const float bv = bias[n];
        const int dv = nt * 16 + ln;             // within-head dim
#pragma unroll
        for (int r = 0; r < 8; ++r) {
            float val = acc[nt][r] + bv;
            if (MODE == 0) val *= 0.125f;        // fold 1/sqrt(HD) into Q
            const int gm = gm0 + r + hi * 8;
            const int b  = gm >> 11;             // /2048
            const int s  = gm & 2047;
            const __bf16 h16 = (__bf16)val;
            if (MODE == 2) {
                O[(((size_t)(b * NH + ng)) * HD + dv) * SS + s] = h16;   // Vt
            } else {
                O[(((size_t)(b * NH + ng)) * SS + s) * HD + dv] = h16;   // Qh/Kh
            }
        }
    }
}

// ---------------------------------------------------------------------------
// Stage 2: flash attention. One wave = one (b, h, 16-query tile); streams keys
// in tiles of 64 with online softmax; O accumulated in f32 (4 x v8f).
// Per 64-key tile: 8 WMMA (S) + 8 WMMA (O), DPP-based row reductions.
// ---------------------------------------------------------------------------
__launch_bounds__(256)
__global__ void attn_kernel(const __bf16* __restrict__ Qh,   // [b][h][s][d], pre-scaled
                            const __bf16* __restrict__ Kh,   // [b][h][s][d]
                            const __bf16* __restrict__ Vt,   // [b][h][dv][s]
                            float* __restrict__ out) {       // [b][s][h*64+dv]
    __shared__ __bf16 Pls[8][16][64];   // per-wave P staging (C-layout -> A-frag)

    const int lane = threadIdx.x & 31;
    const int wv   = threadIdx.x >> 5;
    const int w    = blockIdx.x * 8 + wv;   // 4096 waves
    const int qt   = w & 127;
    const int h    = (w >> 7) & 15;
    const int b    = w >> 11;
    const int ln   = lane & 15;
    const int hi   = lane >> 4;

    const __bf16* Qbase = Qh + (((size_t)(b * NH + h)) * SS + qt * 16 + ln) * HD;
    const __bf16* Kbase = Kh + ((size_t)(b * NH + h)) * SS * HD;
    const __bf16* Vbase = Vt + ((size_t)(b * NH + h)) * HD * SS;

    // Q A-fragments (two 32-wide k chunks of head_dim=64), loaded once.
    BFrag qa[2];
#pragma unroll
    for (int c = 0; c < 2; ++c) {
        qa[c].q[0] = *(const uint4*)(Qbase + c * 32 + hi * 16);
        qa[c].q[1] = *(const uint4*)(Qbase + c * 32 + hi * 16 + 8);
    }

    v8f o[4];
#pragma unroll
    for (int t = 0; t < 4; ++t) o[t] = (v8f){0.f,0.f,0.f,0.f,0.f,0.f,0.f,0.f};
    float mi[8], li[8];
#pragma unroll
    for (int r = 0; r < 8; ++r) { mi[r] = -__builtin_inff(); li[r] = 0.f; }

    for (int kb = 0; kb < SS; kb += 64) {
        // Prefetch next K/V tiles (one 128B line per lane covers each 8KB tile).
        if (kb + 64 < SS) {
            __builtin_prefetch(Kbase + (size_t)(kb + 64 + lane) * HD, 0, 0);
            __builtin_prefetch(Kbase + (size_t)(kb + 96 + lane) * HD, 0, 0);
            __builtin_prefetch(Vbase + (size_t)lane * SS + kb + 64, 0, 0);
            __builtin_prefetch(Vbase + (size_t)(lane + 32) * SS + kb + 64, 0, 0);
        }

        // ---- S tile (16 queries x 64 keys), four 16-col C fragments ----
        v8f s[4];
#pragma unroll
        for (int f = 0; f < 4; ++f) {
            const __bf16* kp = Kbase + (size_t)(kb + f * 16 + ln) * HD + hi * 16;
            BFrag k0, k1;                       // d chunks 0..31 and 32..63
            k0.q[0] = *(const uint4*)kp;        k0.q[1] = *(const uint4*)(kp + 8);
            k1.q[0] = *(const uint4*)(kp + 32); k1.q[1] = *(const uint4*)(kp + 40);
            v8f z = (v8f){0.f,0.f,0.f,0.f,0.f,0.f,0.f,0.f};
            z = wmma_bf16(qa[0].v, k0.v, z);
            s[f] = wmma_bf16(qa[1].v, k1.v, z);
        }

        // ---- online softmax (row stats per lane, rows = vgpr idx + hi*8) ----
        float alpha[8], ps[8];
#pragma unroll
        for (int r = 0; r < 8; ++r) {
            float tm = fmaxf(fmaxf(s[0][r], s[1][r]), fmaxf(s[2][r], s[3][r]));
            tm = red_max16(tm);                 // 4 DPP VALU ops, no LDS
            const float mn = fmaxf(mi[r], tm);
            alpha[r] = __builtin_amdgcn_exp2f((mi[r] - mn) * LOG2E);
            mi[r] = mn;
            ps[r] = 0.f;
        }
#pragma unroll
        for (int f = 0; f < 4; ++f)
#pragma unroll
            for (int r = 0; r < 8; ++r) {
                const float p = __builtin_amdgcn_exp2f((s[f][r] - mi[r]) * LOG2E);
                ps[r] += p;
                Pls[wv][r + hi * 8][f * 16 + ln] = (__bf16)p;  // C-layout -> [m][n]
            }
#pragma unroll
        for (int r = 0; r < 8; ++r) li[r] = li[r] * alpha[r] + red_sum16(ps[r]);
#pragma unroll
        for (int t = 0; t < 4; ++t)
#pragma unroll
            for (int r = 0; r < 8; ++r) o[t][r] *= alpha[r];

        // ---- P back as A-fragments (same-wave DS ops are in-order) ----
        BFrag pa[2];
#pragma unroll
        for (int kc = 0; kc < 2; ++kc) {
            pa[kc].q[0] = *(const uint4*)&Pls[wv][ln][kc * 32 + hi * 16];
            pa[kc].q[1] = *(const uint4*)&Pls[wv][ln][kc * 32 + hi * 16 + 8];
        }

        // ---- O += P @ V : 4 dv-subtiles x 2 key chunks ----
#pragma unroll
        for (int dt = 0; dt < 4; ++dt) {
            const __bf16* vp = Vbase + (size_t)(dt * 16 + ln) * SS + kb + hi * 16;
#pragma unroll
            for (int kc = 0; kc < 2; ++kc) {
                BFrag vb;
                vb.q[0] = *(const uint4*)(vp + kc * 32);
                vb.q[1] = *(const uint4*)(vp + kc * 32 + 8);
                o[dt] = wmma_bf16(pa[kc].v, vb.v, o[dt]);
            }
        }
    }

    // ---- normalize and store f32 output [b][s][h*64+dv] ----
    const int s0 = qt * 16;
#pragma unroll
    for (int dt = 0; dt < 4; ++dt)
#pragma unroll
        for (int r = 0; r < 8; ++r) {
            const float val = o[dt][r] / li[r];
            const int srow = s0 + r + hi * 8;
            out[((size_t)b * SS + srow) * DD + h * HD + dt * 16 + ln] = val;
        }
}

// ---------------------------------------------------------------------------
// Launcher. Workspace: Qh(8MB) + Kh(8MB) + Vt(8MB) bf16 = 24 MB.
// ---------------------------------------------------------------------------
extern "C" void kernel_launch(void* const* d_in, const int* in_sizes, int n_in,
                              void* d_out, int out_size, void* d_ws, size_t ws_size,
                              hipStream_t stream) {
    const float* q  = (const float*)d_in[0];
    const float* k  = (const float*)d_in[1];
    const float* v  = (const float*)d_in[2];
    const float* Wk = (const float*)d_in[3];
    const float* bk = (const float*)d_in[4];
    const float* Wv = (const float*)d_in[5];
    const float* bv = (const float*)d_in[6];

    const size_t PER = (size_t)BB * NH * SS * HD;   // 4M bf16 elems = 8 MB
    __bf16* Qh = (__bf16*)d_ws;
    __bf16* Kh = Qh + PER;
    __bf16* Vt = Kh + PER;

    dim3 blk(256);
    proj_kernel<0><<<512, blk, 0, stream>>>(q, Wk, bk, Qh);   // Q (scaled)
    proj_kernel<1><<<512, blk, 0, stream>>>(k, Wv, bv, Kh);   // K (via Wv, per ref)
    proj_kernel<2><<<512, blk, 0, stream>>>(v, Wv, bv, Vt);   // V transposed
    attn_kernel<<<512, blk, 0, stream>>>(Qh, Kh, Vt, (float*)d_out);
}
